// PaiNN_65876208386292
// MI455X (gfx1250) — compile-verified, matrix operand order checked by reading, not run on hardware
//
#include <hip/hip_runtime.h>
#include <math.h>

// ---------------------------------------------------------------------------
// PaiNN energy + forces for MI455X (gfx1250).  All GEMMs run on
// v_wmma_f32_16x16x32_bf16 (bf16 inputs, f32 accumulate).  Each wave computes
// a 16x64 output strip (4 col tiles share one A fragment); K is a template
// parameter so the K-loop fully unrolls.
// ---------------------------------------------------------------------------

typedef __attribute__((ext_vector_type(16))) __bf16 v16bf;
typedef __attribute__((ext_vector_type(8)))  float  v8f;
typedef __bf16 bf16_t;

#define DEVINL __device__ __forceinline__

constexpr int   N_ = 20000;
constexpr int   E_ = 320000;
constexpr int   H_ = 128;
constexpr int   G_ = 128;
constexpr int   L_ = 3;
constexpr int   B_ = 16;
constexpr float CUT_   = 12.0f;
constexpr float STEP_  = CUT_ / (float)(G_ - 1);
constexpr float COEFF_ = -0.5f / (STEP_ * STEP_);
constexpr float INV3_  = 0.5773502691896258f;   // 1/sqrt(3)
constexpr float INVH_  = 0.08838834764831845f;  // 1/sqrt(128)
constexpr float INV2_  = 0.7071067811865476f;   // 1/sqrt(2)

DEVINL float dev_sig(float x)   { return 1.f / (1.f + __expf(-x)); }
DEVINL float dev_silu(float x)  { return x * dev_sig(x); }
DEVINL float dev_dsilu(float p) { float s = dev_sig(p); return s * (1.f + p * (1.f - s)); }

DEVINL float reduce128(float v, float* sm) {
    int t = threadIdx.x;
    sm[t] = v; __syncthreads();
    #pragma unroll
    for (int s = 64; s > 0; s >>= 1) { if (t < s) sm[t] += sm[t + s]; __syncthreads(); }
    float r = sm[0]; __syncthreads();
    return r;
}
DEVINL float reduce64(float v, float* sm) {
    int t = threadIdx.x;
    sm[t] = v; __syncthreads();
    #pragma unroll
    for (int s = 32; s > 0; s >>= 1) { if (t < s) sm[t] += sm[t + s]; __syncthreads(); }
    float r = sm[0]; __syncthreads();
    return r;
}

// ---------------------------------------------------------------------------
// WMMA GEMM:  C[M,Nout] = act( A[M,TK] * Wt[Nout,TK]^T + bias )
//   A: f32 row-major (converted to bf16 in-register, 16-bit A-fragment layout)
//   Wt: bf16 row-major [Nout,TK] (transposed weight fwd / raw weight for
//       backward data-grad dA = dC * W^T)
//   One wave -> 16 rows x 64 cols (4 col tiles reuse the A fragment).
//   mode 0: none   mode 1: silu (+optional preact save)
//   mode 2: multiply output by dsilu(dpre)
//   addto : C += result
// Requires M%16==0, TK%32==0, Nout%64==0.
// ---------------------------------------------------------------------------
template<int TK>
__global__ void k_wmma_gemm(const float* __restrict__ A, const bf16_t* __restrict__ Wt,
                            const float* __restrict__ bias, float* __restrict__ C,
                            float* __restrict__ preact, const float* __restrict__ dpre,
                            int M, int Nout, int mode, int addto) {
    int lane = threadIdx.x;                 // 0..31
    int row0 = blockIdx.x * 16;
    int col0 = blockIdx.y * 64;
    int mrow = lane & 15;
    int koff = (lane >> 4) * 8;

    const float* arow = A + (size_t)(row0 + mrow) * TK + koff;
    const bf16_t* brow[4];
    #pragma unroll
    for (int c = 0; c < 4; ++c)
        brow[c] = Wt + (size_t)(col0 + c * 16 + mrow) * TK + koff;

    v8f acc[4] = {{}, {}, {}, {}};

    #pragma unroll
    for (int kk = 0; kk < TK; kk += 32) {
        if (kk + 32 < TK) __builtin_prefetch(arow + kk + 32, 0, 1);
        v16bf af;
        #pragma unroll
        for (int t = 0; t < 8; ++t) {
            af[t]     = (__bf16)arow[kk + t];
            af[t + 8] = (__bf16)arow[kk + 16 + t];
        }
        #pragma unroll
        for (int c = 0; c < 4; ++c) {
            v16bf bg;
            #pragma unroll
            for (int t = 0; t < 8; ++t) {
                bg[t]     = brow[c][kk + t];
                bg[t + 8] = brow[c][kk + 16 + t];
            }
            acc[c] = __builtin_amdgcn_wmma_f32_16x16x32_bf16(false, af, false, bg,
                                                             (short)0, acc[c], false, false);
        }
    }

    int rbase = row0 + ((lane >> 4) ? 8 : 0);
    #pragma unroll
    for (int c = 0; c < 4; ++c) {
        int   ncol = col0 + c * 16 + mrow;
        float bval = bias ? bias[ncol] : 0.f;
        #pragma unroll
        for (int r = 0; r < 8; ++r) {
            size_t idx = (size_t)(rbase + r) * Nout + ncol;
            float v = acc[c][r] + bval;
            if (mode == 1) {
                if (preact) preact[idx] = v;
                v = dev_silu(v);
            } else if (mode == 2) {
                v = v * dev_dsilu(dpre[idx]);
            }
            if (addto) C[idx] += v; else C[idx] = v;
        }
    }
}

// ---------------------------------------------------------------------------
// Elementwise helpers
// ---------------------------------------------------------------------------
__global__ void k_fill(float* p, float v, size_t n) {
    size_t i = (size_t)blockIdx.x * blockDim.x + threadIdx.x;
    if (i < n) p[i] = v;
}
__global__ void k_copy(float* d, const float* s, size_t n) {
    size_t i = (size_t)blockIdx.x * blockDim.x + threadIdx.x;
    if (i < n) d[i] = s[i];
}
__global__ void k_scale(float* p, float s, size_t n) {
    size_t i = (size_t)blockIdx.x * blockDim.x + threadIdx.x;
    if (i < n) p[i] *= s;
}

// weight prep: W [Kin,Kout] f32  ->  Wd bf16 [Kin,Kout],  Wt bf16 [Kout,Kin]
__global__ void k_prep_weight(const float* __restrict__ W, bf16_t* __restrict__ Wd,
                              bf16_t* __restrict__ Wt, int Kin, int Kout) {
    int idx = blockIdx.x * blockDim.x + threadIdx.x;
    if (idx >= Kin * Kout) return;
    int k = idx / Kout, c = idx % Kout;
    float v = W[idx];
    Wd[idx] = (__bf16)v;
    Wt[(size_t)c * Kin + k] = (__bf16)v;
}

// ---------------------------------------------------------------------------
// Geometry / embedding
// ---------------------------------------------------------------------------
__global__ void k_edge_geom(const float* __restrict__ pos, const int* __restrict__ ei,
                            float* __restrict__ rij, float* __restrict__ dist,
                            float* __restrict__ rbf) {
    int e = blockIdx.x, t = threadIdx.x;
    __shared__ float sr[3];
    __shared__ float sd;
    int s = ei[e], d = ei[E_ + e];
    if (t < 3) sr[t] = pos[d * 3 + t] - pos[s * 3 + t];
    __syncthreads();
    if (t == 0) {
        float dd = sqrtf(sr[0]*sr[0] + sr[1]*sr[1] + sr[2]*sr[2] + 1e-12f);
        sd = dd; dist[e] = dd;
        rij[e*3+0] = sr[0]; rij[e*3+1] = sr[1]; rij[e*3+2] = sr[2];
    }
    __syncthreads();
    float delta = sd - (float)t * STEP_;
    rbf[(size_t)e * G_ + t] = expf(COEFF_ * delta * delta);
}

__global__ void k_embed(const float* __restrict__ emb, const int* __restrict__ z,
                        float* __restrict__ X0, float* __restrict__ V0) {
    int n = blockIdx.x, h = threadIdx.x;
    X0[(size_t)n * H_ + h] = emb[(size_t)z[n] * H_ + h];
    #pragma unroll
    for (int zz = 0; zz < 3; ++zz) V0[((size_t)n * 3 + zz) * H_ + h] = 0.f;
}

// ---------------------------------------------------------------------------
// LayerNorm fwd / bwd
// ---------------------------------------------------------------------------
__global__ void k_ln_fwd(const float* __restrict__ x, const float* __restrict__ g,
                         const float* __restrict__ b, float* __restrict__ xn,
                         float* __restrict__ mu_o, float* __restrict__ rstd_o) {
    __shared__ float sm[128];
    int n = blockIdx.x, h = threadIdx.x;
    float v  = x[(size_t)n * H_ + h];
    float mu = reduce128(v, sm) * (1.f / H_);
    float dv = v - mu;
    float var = reduce128(dv * dv, sm) * (1.f / H_);
    float rstd = rsqrtf(var + 1e-5f);
    xn[(size_t)n * H_ + h] = dv * rstd * g[h] + b[h];
    if (h == 0) { mu_o[n] = mu; rstd_o[n] = rstd; }
}

__global__ void k_ln_bwd(const float* __restrict__ x0, const float* __restrict__ mu,
                         const float* __restrict__ rstd, const float* __restrict__ gamma,
                         const float* __restrict__ gxn, float* __restrict__ gx) {
    __shared__ float sm[128];
    int n = blockIdx.x, h = threadIdx.x;
    float xhat = (x0[(size_t)n * H_ + h] - mu[n]) * rstd[n];
    float gh   = gxn[(size_t)n * H_ + h] * gamma[h];
    float s1 = reduce128(gh, sm) * (1.f / H_);
    float s2 = reduce128(gh * xhat, sm) * (1.f / H_);
    gx[(size_t)n * H_ + h] += rstd[n] * (gh - s1 - xhat * s2);
}

// ---------------------------------------------------------------------------
// Edge message passing fwd / bwd
// ---------------------------------------------------------------------------
__global__ void k_edge_msg_fwd(const int* __restrict__ ei, const float* __restrict__ XH,
                               const float* __restrict__ RBFH, const float* __restrict__ V0,
                               const float* __restrict__ RIJ,
                               float* __restrict__ ACCX, float* __restrict__ ACCV) {
    int e = blockIdx.x, h = threadIdx.x;
    int s = ei[e], d = ei[E_ + e];
    size_t eb = (size_t)e * 384, sb = (size_t)s * 384;
    float m1 = XH[sb + h]       * RBFH[eb + h];
    float m2 = XH[sb + 128 + h] * RBFH[eb + 128 + h] * INV3_;
    float m3 = XH[sb + 256 + h] * RBFH[eb + 256 + h];
    atomicAdd(&ACCX[(size_t)d * H_ + h], m1);
    #pragma unroll
    for (int zz = 0; zz < 3; ++zz) {
        float v = (V0[((size_t)s * 3 + zz) * H_ + h] * m2 + m3 * RIJ[e * 3 + zz]) * INVH_;
        atomicAdd(&ACCV[((size_t)d * 3 + zz) * H_ + h], v);
    }
}

__global__ void k_edge_msg_bwd(const int* __restrict__ ei, const float* __restrict__ XH,
                               const float* __restrict__ RBFH, const float* __restrict__ V0,
                               const float* __restrict__ RIJ, const float* __restrict__ GDX,
                               const float* __restrict__ GVR, float* __restrict__ GRBFH,
                               float* __restrict__ GXH, float* __restrict__ GVEC,
                               float* __restrict__ GRIJ) {
    __shared__ float sm[128];
    int e = blockIdx.x, h = threadIdx.x;
    int s = ei[e], d = ei[E_ + e];
    size_t eb = (size_t)e * 384, sb = (size_t)s * 384;
    float xh1 = XH[sb + h], xh2 = XH[sb + 128 + h], xh3 = XH[sb + 256 + h];
    float rb1 = RBFH[eb + h], rb2 = RBFH[eb + 128 + h], rb3 = RBFH[eb + 256 + h];
    float m2 = xh2 * rb2 * INV3_;
    float m3 = xh3 * rb3;
    float gm1 = GDX[(size_t)d * H_ + h];
    float gvm[3], v0[3], rr[3];
    #pragma unroll
    for (int zz = 0; zz < 3; ++zz) {
        gvm[zz] = GVR[((size_t)d * 3 + zz) * H_ + h];
        v0[zz]  = V0[((size_t)s * 3 + zz) * H_ + h];
        rr[zz]  = RIJ[e * 3 + zz];
    }
    float gm2 = INVH_ * (gvm[0]*v0[0] + gvm[1]*v0[1] + gvm[2]*v0[2]); // wrt m2(final)
    float gm3 = INVH_ * (gvm[0]*rr[0] + gvm[1]*rr[1] + gvm[2]*rr[2]);
    GRBFH[eb + h]       = gm1 * xh1;
    GRBFH[eb + 128 + h] = gm2 * INV3_ * xh2;
    GRBFH[eb + 256 + h] = gm3 * xh3;
    atomicAdd(&GXH[sb + h],       gm1 * rb1);
    atomicAdd(&GXH[sb + 128 + h], gm2 * INV3_ * rb2);
    atomicAdd(&GXH[sb + 256 + h], gm3 * rb3);
    #pragma unroll
    for (int zz = 0; zz < 3; ++zz)
        atomicAdd(&GVEC[((size_t)s * 3 + zz) * H_ + h], gvm[zz] * m2 * INVH_);
    for (int zz = 0; zz < 3; ++zz) {
        float rsum = reduce128(gvm[zz] * m3 * INVH_, sm);
        if (h == 0) GRIJ[e * 3 + zz] += rsum;
    }
}

__global__ void k_edge_dist_bwd(const float* __restrict__ DIST, const float* __restrict__ RIJ,
                                const float* __restrict__ GRBF, float* __restrict__ GRIJ) {
    __shared__ float sm[128];
    int e = blockIdx.x, h = threadIdx.x;
    float dist = DIST[e];
    float delta = dist - (float)h * STEP_;
    float rb = expf(COEFF_ * delta * delta);
    float v = GRBF[(size_t)e * G_ + h] * rb * 2.f * COEFF_ * delta;
    float gdist = reduce128(v, sm);
    if (h < 3) GRIJ[e * 3 + h] += gdist * RIJ[e * 3 + h] / dist;
}

// ---------------------------------------------------------------------------
// Update-block kernels
// ---------------------------------------------------------------------------
__global__ void k_vdotnorm(const float* __restrict__ VP, const float* __restrict__ X1,
                           float* __restrict__ VDOT, float* __restrict__ VNORM,
                           float* __restrict__ U) {
    int n = blockIdx.x, h = threadIdx.x;
    float ds = 0.f, s2 = 0.f;
    #pragma unroll
    for (int zz = 0; zz < 3; ++zz) {
        float v1 = VP[((size_t)n * 3 + zz) * 256 + h];
        float v2 = VP[((size_t)n * 3 + zz) * 256 + 128 + h];
        ds += v1 * v2; s2 += v2 * v2;
    }
    float vn = sqrtf(s2 + 1e-8f);
    VDOT[(size_t)n * H_ + h] = ds * INVH_;
    VNORM[(size_t)n * H_ + h] = vn;
    U[(size_t)n * 256 + h] = X1[(size_t)n * H_ + h];
    U[(size_t)n * 256 + 128 + h] = vn;
}

__global__ void k_combine(const float* __restrict__ XV, const float* __restrict__ VDOT,
                          const float* __restrict__ X1, const float* __restrict__ V1,
                          const float* __restrict__ VP, float* __restrict__ X2,
                          float* __restrict__ V2) {
    int n = blockIdx.x, h = threadIdx.x;
    size_t nb = (size_t)n * 384;
    float a = XV[nb + h], bb = XV[nb + 128 + h], c = XV[nb + 256 + h];
    X2[(size_t)n * H_ + h] = X1[(size_t)n * H_ + h] + (a + bb * VDOT[(size_t)n * H_ + h]) * INV2_;
    #pragma unroll
    for (int zz = 0; zz < 3; ++zz) {
        size_t vi = ((size_t)n * 3 + zz) * H_ + h;
        V2[vi] = V1[vi] + c * VP[((size_t)n * 3 + zz) * 256 + h];
    }
}

__global__ void k_upd_bwd_a(const float* __restrict__ GX, const float* __restrict__ GVEC,
                            const float* __restrict__ VP, const float* __restrict__ XV,
                            const float* __restrict__ VDOT, float* __restrict__ GXV,
                            float* __restrict__ GVDOT) {
    int n = blockIdx.x, h = threadIdx.x;
    size_t nb = (size_t)n * 384;
    float gx = GX[(size_t)n * H_ + h];
    float bb = XV[nb + 128 + h];
    GXV[nb + h]       = gx * INV2_;
    GXV[nb + 128 + h] = gx * VDOT[(size_t)n * H_ + h] * INV2_;
    float gc = 0.f;
    #pragma unroll
    for (int zz = 0; zz < 3; ++zz)
        gc += GVEC[((size_t)n * 3 + zz) * H_ + h] * VP[((size_t)n * 3 + zz) * 256 + h];
    GXV[nb + 256 + h] = gc;
    GVDOT[(size_t)n * H_ + h] = gx * bb * INV2_;
}

__global__ void k_gu_split(float* __restrict__ GX, const float* __restrict__ GU) {
    int n = blockIdx.x, h = threadIdx.x;
    GX[(size_t)n * H_ + h] += GU[(size_t)n * 256 + h];
}

__global__ void k_vec_bwd(const float* __restrict__ GVEC, const float* __restrict__ GVDOT,
                          const float* __restrict__ GU, const float* __restrict__ VNORM,
                          const float* __restrict__ XV, const float* __restrict__ VP,
                          float* __restrict__ GVP) {
    int n = blockIdx.x, h = threadIdx.x;
    float gd = GVDOT[(size_t)n * H_ + h] * INVH_;
    float gn = GU[(size_t)n * 256 + 128 + h] / VNORM[(size_t)n * H_ + h];
    float c  = XV[(size_t)n * 384 + 256 + h];
    #pragma unroll
    for (int zz = 0; zz < 3; ++zz) {
        float v1 = VP[((size_t)n * 3 + zz) * 256 + h];
        float v2 = VP[((size_t)n * 3 + zz) * 256 + 128 + h];
        float gv = GVEC[((size_t)n * 3 + zz) * H_ + h];
        GVP[((size_t)n * 3 + zz) * 256 + h]       = gv * c + gd * v2;
        GVP[((size_t)n * 3 + zz) * 256 + 128 + h] = gd * v1 + gn * v2;
    }
}

// ---------------------------------------------------------------------------
// Output head + energy + seeds
// ---------------------------------------------------------------------------
__global__ void k_out_reduce(const float* __restrict__ HO, const float* __restrict__ OW2,
                             const float* __restrict__ OB2, const int* __restrict__ batch,
                             float* __restrict__ esum, float* __restrict__ cnt) {
    __shared__ float sm[64];
    int n = blockIdx.x, t = threadIdx.x;
    float pa = reduce64(HO[(size_t)n * 64 + t] * OW2[t], sm);
    if (t == 0) {
        int b = batch[n];
        atomicAdd(&esum[b], pa + OB2[0]);
        atomicAdd(&cnt[b], 1.f);
    }
}

__global__ void k_energy_final(const float* __restrict__ esum, const float* __restrict__ cnt,
                               float* __restrict__ Eout) {
    int t = threadIdx.x;
    if (t < B_) Eout[t] = esum[t] / fmaxf(cnt[t], 1.f);
}

__global__ void k_seed_gpa(const int* __restrict__ batch, const float* __restrict__ cnt,
                           float* __restrict__ gpa) {
    int n = blockIdx.x * blockDim.x + threadIdx.x;
    if (n < N_) gpa[n] = 1.f / fmaxf(cnt[batch[n]], 1.f);
}

__global__ void k_out_bwd(const float* __restrict__ gpa, const float* __restrict__ OW2,
                          const float* __restrict__ PO, float* __restrict__ GPO) {
    int n = blockIdx.x, t = threadIdx.x;
    float p = PO[(size_t)n * 64 + t];
    GPO[(size_t)n * 64 + t] = gpa[n] * OW2[t] * dev_dsilu(p);
}

__global__ void k_force_scatter(const int* __restrict__ ei, const float* __restrict__ GRIJ,
                                float* __restrict__ F) {
    int i = blockIdx.x * blockDim.x + threadIdx.x;
    if (i >= E_ * 3) return;
    int e = i / 3, zz = i % 3;
    int s = ei[e], d = ei[E_ + e];
    float g = GRIJ[i];
    atomicAdd(&F[d * 3 + zz], -g);   // forces = -grad(pos); grad[dst] += g
    atomicAdd(&F[s * 3 + zz],  g);   // grad[src] -= g
}

// ---------------------------------------------------------------------------
// Host launcher
// ---------------------------------------------------------------------------
extern "C" void kernel_launch(void* const* d_in, const int* in_sizes, int n_in,
                              void* d_out, int out_size, void* d_ws, size_t ws_size,
                              hipStream_t stream) {
    const float* pos    = (const float*)d_in[0];
    const float* emb    = (const float*)d_in[1];
    const float* msgW1  = (const float*)d_in[2];
    const float* msgb1  = (const float*)d_in[3];
    const float* msgW2  = (const float*)d_in[4];
    const float* msgb2  = (const float*)d_in[5];
    const float* rbfW   = (const float*)d_in[6];
    const float* rbfb   = (const float*)d_in[7];
    const float* lng    = (const float*)d_in[8];
    const float* lnb    = (const float*)d_in[9];
    const float* vecW   = (const float*)d_in[10];
    const float* updW1  = (const float*)d_in[11];
    const float* updb1  = (const float*)d_in[12];
    const float* updW2  = (const float*)d_in[13];
    const float* updb2  = (const float*)d_in[14];
    const float* outW1  = (const float*)d_in[15];
    const float* outb1  = (const float*)d_in[16];
    const float* outW2  = (const float*)d_in[17];
    const float* outb2  = (const float*)d_in[18];
    const int*   zarr   = (const int*)d_in[19];
    const int*   batch  = (const int*)d_in[20];
    const int*   ei     = (const int*)d_in[21];

    float* Eout = (float*)d_out;
    float* Fout = Eout + B_;

    // ---- workspace bump allocator ----
    char* cur = (char*)d_ws;
    auto allocf = [&](size_t n) -> float*  { float*  p = (float*)cur;  cur += ((n*4 + 255)/256)*256; return p; };
    auto allocb = [&](size_t n) -> bf16_t* { bf16_t* p = (bf16_t*)cur; cur += ((n*2 + 255)/256)*256; return p; };

    const size_t NH = (size_t)N_*H_, NV = (size_t)N_*3*H_, N3H = (size_t)N_*384;
    const size_t N2H = (size_t)N_*256, NVP = (size_t)N_*3*256;

    float* X[L_+1];  float* VEC[L_+1];
    for (int i = 0; i <= L_; ++i) { X[i] = allocf(NH); VEC[i] = allocf(NV); }
    float *X1s[L_], *V1s[L_], *P1s[L_], *XHs[L_], *VPs[L_], *VDs[L_], *VNs[L_], *P2s[L_], *XVs[L_];
    for (int i = 0; i < L_; ++i) {
        X1s[i] = allocf(NH);  V1s[i] = allocf(NV);  P1s[i] = allocf(NH);
        XHs[i] = allocf(N3H); VPs[i] = allocf(NVP); VDs[i] = allocf(NH);
        VNs[i] = allocf(NH);  P2s[i] = allocf(NH);  XVs[i] = allocf(N3H);
    }
    float* MUs  = allocf((size_t)L_*N_);
    float* RSTD = allocf((size_t)L_*N_);
    float* XN   = allocf(NH);
    float* H1   = allocf(NH);
    float* Ubuf = allocf(N2H);
    float* H2   = allocf(NH);
    float* PO   = allocf((size_t)N_*64);
    float* HO   = allocf((size_t)N_*64);
    float* GPO  = allocf((size_t)N_*64);
    float* RIJ  = allocf((size_t)E_*3);
    float* DIST = allocf((size_t)E_);
    float* RBF  = allocf((size_t)E_*G_);
    float* RBFH = allocf((size_t)E_*384);
    float* GRBFH= allocf((size_t)E_*384);
    float* GRBF = allocf((size_t)E_*G_);
    float* GX   = allocf(NH);
    float* GVEC = allocf(NV);
    float* GVR  = allocf(NV);
    float* GXH  = allocf(N3H);
    float* GXV  = allocf(N3H);
    float* GH   = allocf(NH);
    float* GU   = allocf(N2H);
    float* GVD  = allocf(NH);
    float* GVP  = allocf(NVP);
    float* GXN  = allocf(NH);
    float* GRIJ = allocf((size_t)E_*3);
    float* ESUM = allocf(B_);
    float* CNT  = allocf(B_);
    float* GPA  = allocf(N_);

    const size_t sW1 = 128*128, sW2 = 128*384, sWr = 128*384, sWv = 128*256,
                 sU1 = 256*128, sU2 = 128*384, sO1 = 128*64;
    bf16_t *msgW1d = allocb(L_*sW1), *msgW1t = allocb(L_*sW1);
    bf16_t *msgW2d = allocb(L_*sW2), *msgW2t = allocb(L_*sW2);
    bf16_t *rbfWd  = allocb(L_*sWr), *rbfWt  = allocb(L_*sWr);
    bf16_t *vecWd  = allocb(L_*sWv), *vecWt  = allocb(L_*sWv);
    bf16_t *updW1d = allocb(L_*sU1), *updW1t = allocb(L_*sU1);
    bf16_t *updW2d = allocb(L_*sU2), *updW2t = allocb(L_*sU2);
    bf16_t *outW1d = allocb(sO1),    *outW1t = allocb(sO1);

    // ---- launch helpers ----
    auto fill  = [&](float* p, float v, size_t n){ k_fill <<<dim3((unsigned)((n+255)/256)),dim3(256),0,stream>>>(p,v,n); };
    auto copy  = [&](float* d, const float* s, size_t n){ k_copy<<<dim3((unsigned)((n+255)/256)),dim3(256),0,stream>>>(d,s,n); };
    auto scale = [&](float* p, float s, size_t n){ k_scale<<<dim3((unsigned)((n+255)/256)),dim3(256),0,stream>>>(p,s,n); };
    auto gemm  = [&](const float* A, const bf16_t* Wt, const float* bias, float* C,
                     float* preact, const float* dpre, int M, int K, int Nout, int mode, int addto){
        dim3 g(M/16, Nout/64);
        dim3 blk(32);
        switch (K) {
        case 64:  k_wmma_gemm< 64><<<g, blk, 0, stream>>>(A, Wt, bias, C, preact, dpre, M, Nout, mode, addto); break;
        case 128: k_wmma_gemm<128><<<g, blk, 0, stream>>>(A, Wt, bias, C, preact, dpre, M, Nout, mode, addto); break;
        case 256: k_wmma_gemm<256><<<g, blk, 0, stream>>>(A, Wt, bias, C, preact, dpre, M, Nout, mode, addto); break;
        case 384: k_wmma_gemm<384><<<g, blk, 0, stream>>>(A, Wt, bias, C, preact, dpre, M, Nout, mode, addto); break;
        default: break;
        }
    };
    auto prep  = [&](const float* W, bf16_t* Wd, bf16_t* Wt, int Kin, int Kout){
        int n = Kin*Kout;
        k_prep_weight<<<dim3((n+255)/256), dim3(256), 0, stream>>>(W, Wd, Wt, Kin, Kout);
    };

    // ---- weight prep ----
    for (int i = 0; i < L_; ++i) {
        prep(msgW1 + (size_t)i*sW1, msgW1d + (size_t)i*sW1, msgW1t + (size_t)i*sW1, 128, 128);
        prep(msgW2 + (size_t)i*sW2, msgW2d + (size_t)i*sW2, msgW2t + (size_t)i*sW2, 128, 384);
        prep(rbfW  + (size_t)i*sWr, rbfWd  + (size_t)i*sWr, rbfWt  + (size_t)i*sWr, 128, 384);
        prep(vecW  + (size_t)i*sWv, vecWd  + (size_t)i*sWv, vecWt  + (size_t)i*sWv, 128, 256);
        prep(updW1 + (size_t)i*sU1, updW1d + (size_t)i*sU1, updW1t + (size_t)i*sU1, 256, 128);
        prep(updW2 + (size_t)i*sU2, updW2d + (size_t)i*sU2, updW2t + (size_t)i*sU2, 128, 384);
    }
    prep(outW1, outW1d, outW1t, 128, 64);

    // ---- geometry + embedding ----
    k_edge_geom<<<dim3(E_), dim3(G_), 0, stream>>>(pos, ei, RIJ, DIST, RBF);
    k_embed    <<<dim3(N_), dim3(H_), 0, stream>>>(emb, zarr, X[0], VEC[0]);

    // ======================= FORWARD =======================
    for (int i = 0; i < L_; ++i) {
        k_ln_fwd<<<dim3(N_), dim3(H_), 0, stream>>>(X[i], lng + i*H_, lnb + i*H_, XN, MUs + (size_t)i*N_, RSTD + (size_t)i*N_);
        gemm(XN,  msgW1t + (size_t)i*sW1, msgb1 + i*128, H1, P1s[i], nullptr, N_, 128, 128, 1, 0);
        gemm(H1,  msgW2t + (size_t)i*sW2, msgb2 + i*384, XHs[i], nullptr, nullptr, N_, 128, 384, 0, 0);
        gemm(RBF, rbfWt  + (size_t)i*sWr, rbfb  + i*384, RBFH,   nullptr, nullptr, E_, 128, 384, 0, 0);
        copy(X1s[i], X[i], NH);
        copy(V1s[i], VEC[i], NV);
        k_edge_msg_fwd<<<dim3(E_), dim3(H_), 0, stream>>>(ei, XHs[i], RBFH, VEC[i], RIJ, X1s[i], V1s[i]);
        scale(X1s[i], INV2_, NH);
        gemm(V1s[i], vecWt + (size_t)i*sWv, nullptr, VPs[i], nullptr, nullptr, 3*N_, 128, 256, 0, 0);
        k_vdotnorm<<<dim3(N_), dim3(H_), 0, stream>>>(VPs[i], X1s[i], VDs[i], VNs[i], Ubuf);
        gemm(Ubuf, updW1t + (size_t)i*sU1, updb1 + i*128, H2, P2s[i], nullptr, N_, 256, 128, 1, 0);
        gemm(H2,   updW2t + (size_t)i*sU2, updb2 + i*384, XVs[i], nullptr, nullptr, N_, 128, 384, 0, 0);
        k_combine<<<dim3(N_), dim3(H_), 0, stream>>>(XVs[i], VDs[i], X1s[i], V1s[i], VPs[i], X[i+1], VEC[i+1]);
    }

    // output head + energy
    gemm(X[L_], outW1t, outb1, HO, PO, nullptr, N_, 128, 64, 1, 0);
    fill(ESUM, 0.f, B_); fill(CNT, 0.f, B_);
    k_out_reduce<<<dim3(N_), dim3(64), 0, stream>>>(HO, outW2, outb2, batch, ESUM, CNT);
    k_energy_final<<<dim3(1), dim3(32), 0, stream>>>(ESUM, CNT, Eout);

    // ======================= BACKWARD (forces) =======================
    k_seed_gpa<<<dim3((N_+255)/256), dim3(256), 0, stream>>>(batch, CNT, GPA);
    k_out_bwd<<<dim3(N_), dim3(64), 0, stream>>>(GPA, outW2, PO, GPO);
    gemm(GPO, outW1d, nullptr, GX, nullptr, nullptr, N_, 64, 128, 0, 0);
    fill(GVEC, 0.f, NV);
    fill(GRIJ, 0.f, (size_t)E_*3);

    for (int i = L_ - 1; i >= 0; --i) {
        // ---- update block backward ----
        k_upd_bwd_a<<<dim3(N_), dim3(H_), 0, stream>>>(GX, GVEC, VPs[i], XVs[i], VDs[i], GXV, GVD);
        gemm(GXV, updW2d + (size_t)i*sU2, nullptr, GH, nullptr, P2s[i], N_, 384, 128, 2, 0);
        gemm(GH,  updW1d + (size_t)i*sU1, nullptr, GU, nullptr, nullptr, N_, 128, 256, 0, 0);
        k_gu_split<<<dim3(N_), dim3(H_), 0, stream>>>(GX, GU);
        k_vec_bwd<<<dim3(N_), dim3(H_), 0, stream>>>(GVEC, GVD, GU, VNs[i], XVs[i], VPs[i], GVP);
        gemm(GVP, vecWd + (size_t)i*sWv, nullptr, GVEC, nullptr, nullptr, 3*N_, 256, 128, 0, 1);

        // ---- message block backward ----
        scale(GX, INV2_, NH);            // GX now = g_dx = g_x0 (partial)
        copy(GVR, GVEC, NV);             // snapshot g_dvec for dst reads
        gemm(RBF, rbfWt + (size_t)i*sWr, rbfb + i*384, RBFH, nullptr, nullptr, E_, 128, 384, 0, 0); // recompute rbfh
        fill(GXH, 0.f, N3H);
        k_edge_msg_bwd<<<dim3(E_), dim3(H_), 0, stream>>>(ei, XHs[i], RBFH, VEC[i], RIJ, GX, GVR,
                                                          GRBFH, GXH, GVEC, GRIJ);
        gemm(GRBFH, rbfWd + (size_t)i*sWr, nullptr, GRBF, nullptr, nullptr, E_, 384, 128, 0, 0);
        k_edge_dist_bwd<<<dim3(E_), dim3(G_), 0, stream>>>(DIST, RIJ, GRBF, GRIJ);
        gemm(GXH, msgW2d + (size_t)i*sW2, nullptr, GH,  nullptr, P1s[i], N_, 384, 128, 2, 0);
        gemm(GH,  msgW1d + (size_t)i*sW1, nullptr, GXN, nullptr, nullptr, N_, 128, 128, 0, 0);
        k_ln_bwd<<<dim3(N_), dim3(H_), 0, stream>>>(X[i], MUs + (size_t)i*N_, RSTD + (size_t)i*N_,
                                                    lng + i*H_, GXN, GX);
        // GX = g_x0, GVEC = g_vec0 for next (earlier) layer
    }

    // ---- scatter r_ij grads to forces ----
    fill(Fout, 0.f, (size_t)N_*3);
    k_force_scatter<<<dim3((E_*3 + 255)/256), dim3(256), 0, stream>>>(ei, GRIJ, Fout);
}
